// SvmDual_25975962206414
// MI455X (gfx1250) — matrix operand order repaired and checked by reading, not compiled, not run on hardware
//
#include <hip/hip_runtime.h>

// ---------------------------------------------------------------------------
// RBF-kernel SVM predict for MI455X (gfx1250, wave32, WMMA + TDM).
// out[j] = sum_i w_i * exp(-(1/1024) * max(0, nA_i + nB_j - 2 * <xt_i, x_j>))
// GEMM via V_WMMA_F32_16X16X32_BF16 with a bf16 hi/lo split (hi*hi+hi*lo+lo*hi)
// for ~fp32 accuracy. Fast path: x_train pre-converted to bf16 hi/lo in d_ws;
// A tiles streamed into double-buffered LDS with tensor_load_to_lds (TDM,
// hardware row padding), overlapped with WMMA on the previous tile.
// ---------------------------------------------------------------------------

typedef __bf16 bf16_t;
typedef __attribute__((ext_vector_type(16))) __bf16 v16bf;
typedef __attribute__((ext_vector_type(8)))  __bf16 v8bf;
typedef __attribute__((ext_vector_type(8)))  float  v8f;
typedef __attribute__((ext_vector_type(4)))  unsigned int u32x4;
typedef __attribute__((ext_vector_type(8)))  unsigned int u32x8;

#define DD 256            // feature dim (fixed by the reference)
#define LDSTR (DD + 8)    // LDS row stride in bf16 elems (16B pad per 512B row)
#define BN 128            // columns (rows of x) per workgroup
#define BM_T 64           // rows per tile, TDM double-buffered path
#define BM_F 128          // rows per tile, fallback path
#define SPLITY 8
#define KSCALE (-1.0f / 1024.0f)   // -(1/sigma)^2, sigma = 32

// ---- TDM-path LDS byte offsets (dynamic smem, no static LDS in kernel) ----
#define TD_BHI 0
#define TD_BLO (BN * LDSTR * 2)                   // 67584
#define TD_A   (2 * BN * LDSTR * 2)               // 135168
#define TD_ABYTES (BM_T * LDSTR * 2)              // 33792
#define TD_AHI(b) (TD_A + (b) * 2 * TD_ABYTES)
#define TD_ALO(b) (TD_AHI(b) + TD_ABYTES)
#define TD_NB  (TD_A + 4 * TD_ABYTES)             // 270336
#define TD_COL (TD_NB + BN * 4)                   // 270848
#define TD_SMEM (TD_COL + BN * 4)                 // 271360 B < 320 KB WGP LDS

__device__ __forceinline__ v16bf cat16(v8bf a, v8bf b) {
  return __builtin_shufflevector(a, b, 0,1,2,3,4,5,6,7,8,9,10,11,12,13,14,15);
}

// 8 f32 -> bf16 hi/lo; accumulate sum of squares.
__device__ __forceinline__ void cvt8(const float* f, v8bf& h8, v8bf& l8, float& ss) {
#pragma unroll
  for (int i = 0; i < 8; ++i) {
    float v = f[i];
    __bf16 h = (__bf16)v;
    h8[i] = h;
    l8[i] = (__bf16)(v - (float)h);
    ss = fmaf(v, v, ss);
  }
}

// Thread loads 128 contiguous floats of one row -> hi/lo bf16 in LDS.
__device__ __forceinline__ float loadTileCvt(const float* __restrict__ src,
                                             bf16_t* hi, bf16_t* lo,
                                             int r, int cHalf) {
  float ss = 0.f;
#pragma unroll
  for (int c = 0; c < 128; c += 8) {
    float buf[8];
    *(float4*)(buf)     = *(const float4*)(src + c);
    *(float4*)(buf + 4) = *(const float4*)(src + c + 4);
    v8bf h8, l8;
    cvt8(buf, h8, l8, ss);
    *(v8bf*)(hi + (size_t)r * LDSTR + cHalf + c) = h8;
    *(v8bf*)(lo + (size_t)r * LDSTR + cHalf + c) = l8;
  }
  return ss;
}

// ---------------------------------------------------------------------------
// TDM: DMA a dense [rows x 512B] 2D block from global into LDS, with hardware
// padding of 16B after every 512B row (matches LDSTR). D# per CDNA5 ISA ch.8:
// group0 = {count, lds_addr, global_addr, type=2}, group1 = dims/strides/pad.
// 2-operand form => 2D tensor (VADDR2/3 = NULL).
// ---------------------------------------------------------------------------
__device__ __forceinline__ void tdm_load_2d(const void* gsrc, unsigned ldsOff,
                                            unsigned rows) {
  unsigned long long ga = (unsigned long long)gsrc;
  const unsigned dim0 = DD / 2;                 // 128 DWORDs per row (data_size=4B)
  u32x4 g0;
  g0[0] = 1u;                                   // count=1, user descriptor
  g0[1] = ldsOff;                               // lds_addr (bytes)
  g0[2] = (unsigned)ga;                         // global_addr[31:0]
  g0[3] = (unsigned)(ga >> 32) | (2u << 30);    // global_addr[56:32] | type=2
  u32x8 g1;
  g1[0] = (2u << 16)                            // data_size = 4 bytes
        | (1u << 20)                            // pad_enable
        | (6u << 22)                            // pad_interval: 8B<<6 = 512B
        | (3u << 25);                           // pad_amount: (3+1)*4B = 16B
  g1[1] = (dim0 & 0xffffu) << 16;               // tensor_dim0[15:0]
  g1[2] = (dim0 >> 16) | ((rows & 0xffffu) << 16);  // dim0 hi | tensor_dim1 lo
  g1[3] = (rows >> 16) | (dim0 << 16);          // dim1 hi | tile_dim0
  g1[4] = rows;                                 // tile_dim1 (tile_dim2 = 0)
  g1[5] = dim0;                                 // tensor_dim0_stride[31:0]
  g1[6] = 0u;
  g1[7] = 0u;
  asm volatile("tensor_load_to_lds %0, %1" :: "s"(g0), "s"(g1) : "memory");
}

// ---------------------------------------------------------------------------
// Pre-pass: x_train -> bf16 hi/lo copies + per-row {w = alpha*y, ||row||^2}.
// One wave per row (32 lanes x 8 elems = 256).
// ---------------------------------------------------------------------------
extern "C" __global__ __launch_bounds__(256)
void svm_rbf_cvt(const float* __restrict__ xt, const float* __restrict__ yt,
                 const float* __restrict__ al, bf16_t* __restrict__ hi,
                 bf16_t* __restrict__ lo, float2* __restrict__ wn, int N) {
  const int g    = blockIdx.x * blockDim.x + threadIdx.x;
  const int row  = g >> 5;
  const int lane = g & 31;
  if (row >= N) return;
  const float* src = xt + (size_t)row * DD + lane * 8;
  float buf[8];
  *(float4*)(buf)     = *(const float4*)(src);
  *(float4*)(buf + 4) = *(const float4*)(src + 4);
  v8bf h8, l8;
  float ss = 0.f;
  cvt8(buf, h8, l8, ss);
  *(v8bf*)(hi + (size_t)row * DD + lane * 8) = h8;
  *(v8bf*)(lo + (size_t)row * DD + lane * 8) = l8;
#pragma unroll
  for (int d = 16; d > 0; d >>= 1) ss += __shfl_xor(ss, d);
  if (lane == 0) wn[row] = make_float2(al[row] * yt[row], ss);
}

// ---------------------------------------------------------------------------
// Fast path main kernel: TDM-fed, double-buffered A tiles (BM_T=64 rows).
// 8 waves: 2 (M) x 4 (N); each wave computes a 32x32 patch = 2x2 WMMA tiles.
// ---------------------------------------------------------------------------
extern "C" __global__ __launch_bounds__(256)
void svm_rbf_main_tdm(const bf16_t* __restrict__ xtHi, const bf16_t* __restrict__ xtLo,
                      const float2* __restrict__ wnG, const float* __restrict__ xq,
                      float* __restrict__ partial, int N, int M, int rowsPerSplit)
{
  extern __shared__ char smem[];
  bf16_t* Bhi    = (bf16_t*)(smem + TD_BHI);
  bf16_t* Blo    = (bf16_t*)(smem + TD_BLO);
  float*  nB     = (float*)(smem + TD_NB);
  float*  colSum = (float*)(smem + TD_COL);

  const int tid   = threadIdx.x;
  const int lane  = tid & 31;
  const int wid   = tid >> 5;
  const int lhalf = lane >> 4;
  const int lmod  = lane & 15;
  const int mwave = (wid >> 2) * 32;   // 0 / 32
  const int nwave = (wid & 3) * 32;    // 0..96

  const int n0      = blockIdx.x * BN;
  const int i_begin = blockIdx.y * rowsPerSplit;
  const int numTiles = rowsPerSplit / BM_T;

  // Kick off DMA of tile 0 (buffer 0) before doing any VALU work.
  if (wid == 0) {
    tdm_load_2d(xtHi + (size_t)i_begin * DD, TD_AHI(0), BM_T);
    tdm_load_2d(xtLo + (size_t)i_begin * DD, TD_ALO(0), BM_T);
  }

  // Convert the resident B block (this WG's 128 columns of x) while DMA runs.
  {
    const int r = tid >> 1, cHalf = (tid & 1) * 128;
    float ss = loadTileCvt(xq + (size_t)(n0 + r) * DD + cHalf, Bhi, Blo, r, cHalf);
    ss += __shfl_xor(ss, 1);
    if ((tid & 1) == 0) nB[r] = ss;
  }
  if (tid < BN) colSum[tid] = 0.f;
  if (wid == 0) __builtin_amdgcn_s_wait_tensorcnt(0);
  __syncthreads();

  const float nb0 = nB[nwave + lmod];
  const float nb1 = nB[nwave + 16 + lmod];
  float colAcc0 = 0.f, colAcc1 = 0.f;

  for (int t = 0; t < numTiles; ++t) {
    const int i0  = i_begin + t * BM_T;
    const int buf = t & 1;

    // DMA next tile into the other buffer; overlaps with WMMA below.
    if (wid == 0 && t + 1 < numTiles) {
      tdm_load_2d(xtHi + (size_t)(i0 + BM_T) * DD, TD_AHI(buf ^ 1), BM_T);
      tdm_load_2d(xtLo + (size_t)(i0 + BM_T) * DD, TD_ALO(buf ^ 1), BM_T);
    }

    const bf16_t* Ah = (const bf16_t*)(smem + TD_AHI(buf));
    const bf16_t* Al = (const bf16_t*)(smem + TD_ALO(buf));

    v8f acc[2][2];
#pragma unroll
    for (int mi = 0; mi < 2; ++mi)
#pragma unroll
      for (int ni = 0; ni < 2; ++ni)
        acc[mi][ni] = (v8f){};

    for (int ks = 0; ks < DD; ks += 32) {
      v16bf bh[2], bl[2];
#pragma unroll
      for (int ni = 0; ni < 2; ++ni) {
        const size_t ob = (size_t)(nwave + ni * 16 + lmod) * LDSTR + ks + lhalf * 8;
        bh[ni] = cat16(*(const v8bf*)(Bhi + ob), *(const v8bf*)(Bhi + ob + 16));
        bl[ni] = cat16(*(const v8bf*)(Blo + ob), *(const v8bf*)(Blo + ob + 16));
      }
#pragma unroll
      for (int mi = 0; mi < 2; ++mi) {
        const size_t oa = (size_t)(mwave + mi * 16 + lmod) * LDSTR + ks + lhalf * 8;
        v16bf ah  = cat16(*(const v8bf*)(Ah + oa), *(const v8bf*)(Ah + oa + 16));
        v16bf alo = cat16(*(const v8bf*)(Al + oa), *(const v8bf*)(Al + oa + 16));
#pragma unroll
        for (int ni = 0; ni < 2; ++ni) {
          acc[mi][ni] = __builtin_amdgcn_wmma_f32_16x16x32_bf16(
              false, ah,  false, bh[ni], (short)0, acc[mi][ni], false, false);
          acc[mi][ni] = __builtin_amdgcn_wmma_f32_16x16x32_bf16(
              false, ah,  false, bl[ni], (short)0, acc[mi][ni], false, false);
          acc[mi][ni] = __builtin_amdgcn_wmma_f32_16x16x32_bf16(
              false, alo, false, bh[ni], (short)0, acc[mi][ni], false, false);
        }
      }
    }

    // Fused epilogue: sq -> exp -> weighted column accumulation.
#pragma unroll
    for (int mi = 0; mi < 2; ++mi) {
      const int rowb = i0 + mwave + mi * 16 + lhalf * 8;   // C layout: M=lhalf*8+v
      float2 wn[8];
#pragma unroll
      for (int v = 0; v < 8; ++v) wn[v] = wnG[rowb + v];   // L2-resident broadcast
#pragma unroll
      for (int v = 0; v < 8; ++v) {
        float sq0 = fmaxf(wn[v].y + nb0 - 2.0f * acc[mi][0][v], 0.0f);
        float sq1 = fmaxf(wn[v].y + nb1 - 2.0f * acc[mi][1][v], 0.0f);
        colAcc0 += wn[v].x * __expf(KSCALE * sq0);
        colAcc1 += wn[v].x * __expf(KSCALE * sq1);
      }
    }

    if (wid == 0 && t + 1 < numTiles) __builtin_amdgcn_s_wait_tensorcnt(0);
    __syncthreads();
  }

  atomicAdd(&colSum[nwave + lmod],      colAcc0);
  atomicAdd(&colSum[nwave + 16 + lmod], colAcc1);
  __syncthreads();
  if (tid < BN)
    partial[(size_t)blockIdx.y * M + n0 + tid] = colSum[tid];
}

// ---------------------------------------------------------------------------
// Fallback main kernel (no workspace needed): in-kernel conversion, BM_F=128.
// ---------------------------------------------------------------------------
extern "C" __global__ __launch_bounds__(256)
void svm_rbf_main(const float* __restrict__ xt, const float* __restrict__ yt,
                  const float* __restrict__ al, const float* __restrict__ xq,
                  float* __restrict__ partial, int N, int M, int rowsPerSplit)
{
  extern __shared__ char smem[];
  bf16_t* Bhi = (bf16_t*)smem;
  bf16_t* Blo = Bhi + (size_t)BN * LDSTR;
  bf16_t* Ahi = Blo + (size_t)BN * LDSTR;
  bf16_t* Alo = Ahi + (size_t)BM_F * LDSTR;
  float*  nB  = (float*)(Alo + (size_t)BM_F * LDSTR);
  float2* wnA = (float2*)(nB + BN);
  float*  colSum = (float*)(wnA + BM_F);

  const int tid   = threadIdx.x;
  const int lane  = tid & 31;
  const int wid   = tid >> 5;
  const int lhalf = lane >> 4;
  const int lmod  = lane & 15;
  const int mwave = (wid >> 2) * 64;
  const int nwave = (wid & 3) * 32;

  const int n0      = blockIdx.x * BN;
  const int i_begin = blockIdx.y * rowsPerSplit;
  const int r     = tid >> 1;
  const int cHalf = (tid & 1) * 128;

  {
    float ss = loadTileCvt(xq + (size_t)(n0 + r) * DD + cHalf, Bhi, Blo, r, cHalf);
    ss += __shfl_xor(ss, 1);
    if ((tid & 1) == 0) nB[r] = ss;
  }
  if (tid < BN) colSum[tid] = 0.f;
  __syncthreads();

  const float nb0 = nB[nwave + lmod];
  const float nb1 = nB[nwave + 16 + lmod];
  float colAcc0 = 0.f, colAcc1 = 0.f;

  const int numTiles = rowsPerSplit / BM_F;
  for (int t = 0; t < numTiles; ++t) {
    const int i0 = i_begin + t * BM_F;
    __syncthreads();
    {
      const float* src = xt + (size_t)(i0 + r) * DD + cHalf;
      float ss = loadTileCvt(src, Ahi, Alo, r, cHalf);
      ss += __shfl_xor(ss, 1);
      if ((tid & 1) == 0)
        wnA[r] = make_float2(al[i0 + r] * yt[i0 + r], ss);
      if (t + 1 < numTiles)
        __builtin_prefetch(src + (size_t)BM_F * DD, 0, 0);
    }
    __syncthreads();

    v8f acc[4][2];
#pragma unroll
    for (int mi = 0; mi < 4; ++mi)
#pragma unroll
      for (int ni = 0; ni < 2; ++ni)
        acc[mi][ni] = (v8f){};

    for (int ks = 0; ks < DD; ks += 32) {
      v16bf bh[2], bl[2];
#pragma unroll
      for (int ni = 0; ni < 2; ++ni) {
        const size_t ob = (size_t)(nwave + ni * 16 + lmod) * LDSTR + ks + lhalf * 8;
        bh[ni] = cat16(*(const v8bf*)(Bhi + ob), *(const v8bf*)(Bhi + ob + 16));
        bl[ni] = cat16(*(const v8bf*)(Blo + ob), *(const v8bf*)(Blo + ob + 16));
      }
#pragma unroll
      for (int mi = 0; mi < 4; ++mi) {
        const size_t oa = (size_t)(mwave + mi * 16 + lmod) * LDSTR + ks + lhalf * 8;
        v16bf ah  = cat16(*(const v8bf*)(Ahi + oa), *(const v8bf*)(Ahi + oa + 16));
        v16bf alo = cat16(*(const v8bf*)(Alo + oa), *(const v8bf*)(Alo + oa + 16));
#pragma unroll
        for (int ni = 0; ni < 2; ++ni) {
          acc[mi][ni] = __builtin_amdgcn_wmma_f32_16x16x32_bf16(
              false, ah,  false, bh[ni], (short)0, acc[mi][ni], false, false);
          acc[mi][ni] = __builtin_amdgcn_wmma_f32_16x16x32_bf16(
              false, ah,  false, bl[ni], (short)0, acc[mi][ni], false, false);
          acc[mi][ni] = __builtin_amdgcn_wmma_f32_16x16x32_bf16(
              false, alo, false, bh[ni], (short)0, acc[mi][ni], false, false);
        }
      }
    }

#pragma unroll
    for (int mi = 0; mi < 4; ++mi) {
      float2 wn[8];
#pragma unroll
      for (int v = 0; v < 8; ++v)
        wn[v] = wnA[mwave + mi * 16 + lhalf * 8 + v];
#pragma unroll
      for (int v = 0; v < 8; ++v) {
        float sq0 = fmaxf(wn[v].y + nb0 - 2.0f * acc[mi][0][v], 0.0f);
        float sq1 = fmaxf(wn[v].y + nb1 - 2.0f * acc[mi][1][v], 0.0f);
        colAcc0 += wn[v].x * __expf(KSCALE * sq0);
        colAcc1 += wn[v].x * __expf(KSCALE * sq1);
      }
    }
  }

  __syncthreads();
  atomicAdd(&colSum[nwave + lmod],      colAcc0);
  atomicAdd(&colSum[nwave + 16 + lmod], colAcc1);
  __syncthreads();
  if (tid < BN)
    partial[(size_t)blockIdx.y * M + n0 + tid] = colSum[tid];
}

// Deterministic final reduction over the row-split partials.
extern "C" __global__ void svm_rbf_reduce(const float* __restrict__ partial,
                                          float* __restrict__ out, int M, int nsplit) {
  int j = blockIdx.x * blockDim.x + threadIdx.x;
  if (j < M) {
    float s = 0.f;
    for (int y = 0; y < nsplit; ++y) s += partial[(size_t)y * M + j];
    out[j] = s;
  }
}

extern "C" void kernel_launch(void* const* d_in, const int* in_sizes, int n_in,
                              void* d_out, int out_size, void* d_ws, size_t ws_size,
                              hipStream_t stream) {
  const float* xt = (const float*)d_in[0];   // x_train [N, 256]
  const float* yt = (const float*)d_in[1];   // y_train [N]
  const float* al = (const float*)d_in[2];   // alpha   [N]
  const float* xq = (const float*)d_in[3];   // x       [M, 256]
  float* out = (float*)d_out;                // [1,1,M] -> M floats

  const int N = in_sizes[1];
  const int M = out_size;

  // Workspace layout for the TDM path (256B-aligned regions).
  const size_t hiBytes = (size_t)N * DD * sizeof(bf16_t);
  size_t o_xtHi = 0;
  size_t o_xtLo = (o_xtHi + hiBytes + 255) & ~(size_t)255;
  size_t o_wn   = (o_xtLo + hiBytes + 255) & ~(size_t)255;
  size_t o_part = (o_wn + (size_t)N * sizeof(float2) + 255) & ~(size_t)255;

  int splitY = SPLITY;
  while (splitY > 1 &&
         ((N % (splitY * BM_T)) != 0 ||
          o_part + (size_t)splitY * M * sizeof(float) > ws_size))
    splitY >>= 1;

  const bool tdmOK =
      (M % BN == 0) && (N % BM_T == 0) && (N % 8 == 0) &&
      (o_part + (size_t)((splitY > 1) ? splitY : 0) * M * sizeof(float) <= ws_size);

  if (tdmOK) {
    const int rowsPerSplit = N / splitY;
    bf16_t* xtHi = (bf16_t*)((char*)d_ws + o_xtHi);
    bf16_t* xtLo = (bf16_t*)((char*)d_ws + o_xtLo);
    float2* wn   = (float2*)((char*)d_ws + o_wn);
    float* partial = (splitY > 1) ? (float*)((char*)d_ws + o_part) : out;

    svm_rbf_cvt<<<(N * 32 + 255) / 256, 256, 0, stream>>>(xt, yt, al, xtHi, xtLo, wn, N);

    (void)hipFuncSetAttribute((const void*)svm_rbf_main_tdm,
                              hipFuncAttributeMaxDynamicSharedMemorySize, TD_SMEM);
    dim3 grid(M / BN, splitY);
    svm_rbf_main_tdm<<<grid, 256, TD_SMEM, stream>>>(xtHi, xtLo, wn, xq, partial,
                                                     N, M, rowsPerSplit);
    if (splitY > 1)
      svm_rbf_reduce<<<(M + 255) / 256, 256, 0, stream>>>(partial, out, M, splitY);
    return;
  }

  // ---- fallback: self-contained kernel, no workspace conversion ----
  int splitF = SPLITY;
  while (splitF > 1 &&
         ((N % (splitF * BM_F)) != 0 ||
          (size_t)splitF * M * sizeof(float) > ws_size))
    splitF >>= 1;
  const int rowsPerSplit = N / splitF;
  float* partial = (splitF > 1) ? (float*)d_ws : out;

  const size_t smemBytes =
      (size_t)2 * (BN + BM_F) * LDSTR * sizeof(bf16_t)
      + BN * sizeof(float) + BM_F * sizeof(float2) + BN * sizeof(float);

  (void)hipFuncSetAttribute((const void*)svm_rbf_main,
                            hipFuncAttributeMaxDynamicSharedMemorySize,
                            (int)smemBytes);
  dim3 grid(M / BN, splitF);
  svm_rbf_main<<<grid, 256, smemBytes, stream>>>(xt, yt, al, xq, partial,
                                                 N, M, rowsPerSplit);
  if (splitF > 1)
    svm_rbf_reduce<<<(M + 255) / 256, 256, 0, stream>>>(partial, out, M, splitF);
}
